// MultiHeadSelfAttention_71356586656345
// MI455X (gfx1250) — matrix operand order repaired and checked by reading, not compiled, not run on hardware
//
#include <hip/hip_runtime.h>
#include <hip/hip_bf16.h>

// ---------------------------------------------------------------------------
// MHA: x[B,S,E] -> qkv proj -> causal softmax attention -> out proj
// B=4, S=2048, E=2048, H=16, D=128.  All matrix math on v_wmma_f32_16x16x32_bf16.
// GEMMs use double-buffered LDS staging via global_load_async_to_lds_b128
// (ASYNCcnt-tracked DMA, overlapped with WMMA compute).
// ---------------------------------------------------------------------------

typedef __attribute__((ext_vector_type(16))) __bf16 v16bf;
typedef __attribute__((ext_vector_type(8)))  __bf16 v8bf;
typedef __attribute__((ext_vector_type(8)))  float  v8f;

static constexpr int Bc = 4, Sc = 2048, Ec = 2048, Hc = 16, Dc = 128;

__device__ inline v16bf cat16(v8bf lo, v8bf hi) {
  return __builtin_shufflevector(lo, hi, 0,1,2,3,4,5,6,7,8,9,10,11,12,13,14,15);
}

__device__ inline v8f wmma_bf16(v16bf a, v16bf b, v8f c) {
  // D = A(16x32 bf16) * B(32x16 bf16) + C(16x16 f32)
  return __builtin_amdgcn_wmma_f32_16x16x32_bf16(false, a, false, b, (short)0, c, false, false);
}

// Async DMA: global -> LDS, 16B per lane, tracked on ASYNCcnt.
__device__ inline void async_g2l_b128(const __bf16* gaddr, const __bf16* lds_ptr) {
  unsigned lds_off = (unsigned)(unsigned long long)lds_ptr;  // low 32b of shared aperture = LDS offset
  asm volatile("global_load_async_to_lds_b128 %0, %1, off"
               :: "v"(lds_off), "v"(gaddr) : "memory");
}
__device__ inline void wait_async0() {
  asm volatile("s_wait_asynccnt 0x0" ::: "memory");
}

// ---------------------------------------------------------------------------
// fp32 -> bf16 elementwise convert (vector-4 per thread)
// ---------------------------------------------------------------------------
__global__ __launch_bounds__(256) void cvt_f32_bf16(const float* __restrict__ in,
                                                    __bf16* __restrict__ out, int n) {
  int i = (blockIdx.x * blockDim.x + threadIdx.x) * 4;
  if (i + 3 < n) {
    float4 f = *(const float4*)(in + i);
    out[i + 0] = (__bf16)f.x;
    out[i + 1] = (__bf16)f.y;
    out[i + 2] = (__bf16)f.z;
    out[i + 3] = (__bf16)f.w;
  }
}

// ---------------------------------------------------------------------------
// Transpose+convert: W[K][N] f32 -> Wt[N][K] bf16  (32x32 LDS tiles)
// ---------------------------------------------------------------------------
__global__ __launch_bounds__(256) void transpose_cvt(const float* __restrict__ W,
                                                     __bf16* __restrict__ Wt,
                                                     int K, int N) {
  __shared__ float t[32][33];
  int k0 = blockIdx.y * 32, n0 = blockIdx.x * 32;
  int tx = threadIdx.x;
  for (int i = threadIdx.y; i < 32; i += 8)
    t[i][tx] = W[(size_t)(k0 + i) * N + n0 + tx];
  __syncthreads();
  for (int i = threadIdx.y; i < 32; i += 8)
    Wt[(size_t)(n0 + i) * K + k0 + tx] = (__bf16)t[tx][i];
}

// ---------------------------------------------------------------------------
// Tiled bf16 GEMM:  C[M,N] = A[M,K] * Bt[N,K]^T   (Bt stored N-major, K contig)
// Block: 256 thr (8 waves), tile 128x128, BK=64.  Wave grid 2(M) x 4(N),
// each wave: 4x2 WMMA fragments = 64x32 output.
// Double-buffered async-to-LDS staging: tile k+1 DMA overlaps tile k WMMAs.
// MODE 0: scatter epilogue -> Q[B,H,S,D], K[B,H,S,D], Vt[B,H,D,S] (bf16)
// MODE 1: plain f32 row-major C store
// ---------------------------------------------------------------------------
template <int MODE>
__global__ __launch_bounds__(256)
void gemm_bf16(const __bf16* __restrict__ A, const __bf16* __restrict__ Bt,
               float* __restrict__ Cf,
               __bf16* __restrict__ Qb, __bf16* __restrict__ Kb, __bf16* __restrict__ Vtb,
               int M, int N, int K) {
  constexpr int BK = 64, LDT = 72;  // 72*2B = 144B row stride: 16B aligned, conflict-free
  __shared__ __bf16 As[2][128 * LDT];
  __shared__ __bf16 Bs[2][128 * LDT];

  const int tid = threadIdx.x;
  const int wid = tid >> 5, lane = tid & 31;
  const int ln = lane & 15, hl = lane >> 4;
  const int waveM = wid >> 2, waveN = wid & 3;
  const int row0 = blockIdx.y * 128;
  const int col0 = blockIdx.x * 128;

  // this thread's 4 staging segments (16B each per array)
  const int sr[4] = { (tid + 0) >> 3, (tid + 256) >> 3, (tid + 512) >> 3, (tid + 768) >> 3 };
  const int sc = (tid & 7) << 3;

  auto stage = [&](int buf, int k0) {
#pragma unroll
    for (int i = 0; i < 4; ++i) {
      async_g2l_b128(&A [(size_t)(row0 + sr[i]) * K + k0 + sc], &As[buf][sr[i] * LDT + sc]);
      async_g2l_b128(&Bt[(size_t)(col0 + sr[i]) * K + k0 + sc], &Bs[buf][sr[i] * LDT + sc]);
    }
  };

  v8f acc[4][2] = {};
  const int nk = K / BK;

  stage(0, 0);
  wait_async0();
  __syncthreads();

  for (int kt = 0; kt < nk; ++kt) {
    const int buf = kt & 1;
    if (kt + 1 < nk) stage(buf ^ 1, (kt + 1) * BK);   // DMA next tile during compute

#pragma unroll
    for (int kk = 0; kk < BK; kk += 32) {
      v16bf a[4], b[2];
#pragma unroll
      for (int mt = 0; mt < 4; ++mt) {
        int r = waveM * 64 + mt * 16 + ln;
        v8bf lo = *(const v8bf*)&As[buf][r * LDT + kk + 8 * hl];
        v8bf hi = *(const v8bf*)&As[buf][r * LDT + kk + 16 + 8 * hl];
        a[mt] = cat16(lo, hi);
      }
#pragma unroll
      for (int nt = 0; nt < 2; ++nt) {
        int r = waveN * 32 + nt * 16 + ln;
        v8bf lo = *(const v8bf*)&Bs[buf][r * LDT + kk + 8 * hl];
        v8bf hi = *(const v8bf*)&Bs[buf][r * LDT + kk + 16 + 8 * hl];
        b[nt] = cat16(lo, hi);
      }
#pragma unroll
      for (int mt = 0; mt < 4; ++mt)
#pragma unroll
        for (int nt = 0; nt < 2; ++nt)
          acc[mt][nt] = wmma_bf16(a[mt], b[nt], acc[mt][nt]);
    }

    wait_async0();     // own next-tile DMA done (no-op on last iter)
    __syncthreads();   // all waves: staging visible + done reading current buf
  }

  // epilogue: C layout -> lane holds N=ln, rows v + 8*hl
#pragma unroll
  for (int mt = 0; mt < 4; ++mt)
#pragma unroll
    for (int nt = 0; nt < 2; ++nt)
#pragma unroll
      for (int v = 0; v < 8; ++v) {
        int mg = row0 + waveM * 64 + mt * 16 + v + 8 * hl;
        int ng = col0 + waveN * 32 + nt * 16 + ln;
        float val = acc[mt][nt][v];
        if (MODE == 1) {
          Cf[(size_t)mg * N + ng] = val;
        } else {
          int bb = mg >> 11, s = mg & 2047;      // m = b*S + s
          int part = ng >> 11, e = ng & 2047;    // n = part*E + e
          int head = e >> 7, dd = e & 127;
          size_t bh = (size_t)bb * Hc + head;
          __bf16 bv = (__bf16)val;
          if (part == 0)      Qb[(bh * Sc + s) * Dc + dd] = bv;
          else if (part == 1) Kb[(bh * Sc + s) * Dc + dd] = bv;
          else                Vtb[(bh * Dc + dd) * Sc + s] = bv;
        }
      }
}

// ---------------------------------------------------------------------------
// Causal flash attention. Block = 8 waves, each wave owns a 16-row Q tile
// (independent online softmax). Q[B,H,S,D], K[B,H,S,D], Vt[B,H,D,S] bf16.
// Output r[B,S,E] bf16. Logits kept in exp2 domain: scale2 = log2e/sqrt(D).
// ---------------------------------------------------------------------------
__global__ __launch_bounds__(256)
void flash_attn(const __bf16* __restrict__ Qb, const __bf16* __restrict__ Kb,
                const __bf16* __restrict__ Vtb, __bf16* __restrict__ rb) {
  constexpr int PLD = 48;                  // 96B stride: 16B aligned
  __shared__ __bf16 Pl[8][16 * PLD];       // per-wave 16x32 P staging

  const int wid = threadIdx.x >> 5, lane = threadIdx.x & 31;
  const int ln = lane & 15, hl = lane >> 4;
  const int bh = blockIdx.y;               // 0..B*H-1
  const int b = bh >> 4, hd = bh & 15;
  const int q0 = blockIdx.x * 128 + wid * 16;

  const __bf16* Qp = Qb + (size_t)bh * Sc * Dc;
  const __bf16* Kp = Kb + (size_t)bh * Sc * Dc;
  const __bf16* Vp = Vtb + (size_t)bh * Dc * Sc;

  // Q A-fragments: rows q0+ln, 4 k-steps of 32 over D=128
  v16bf qa[4];
  {
    const __bf16* qr = Qp + (size_t)(q0 + ln) * Dc;
#pragma unroll
    for (int kk = 0; kk < 4; ++kk) {
      v8bf lo = *(const v8bf*)&qr[kk * 32 + 8 * hl];
      v8bf hi = *(const v8bf*)&qr[kk * 32 + 16 + 8 * hl];
      qa[kk] = cat16(lo, hi);
    }
  }

  float m_i[8], l_i[8];
  v8f o[8] = {};
#pragma unroll
  for (int v = 0; v < 8; ++v) { m_i[v] = -INFINITY; l_i[v] = 0.0f; }

  const float scale2 = 0.08838834764831845f * 1.4426950408889634f;  // rsqrt(128)*log2e
  __bf16* pl = &Pl[wid][0];
  const int nchunks = (q0 + 47) >> 5;      // causal: keys [0, q0+16)

  for (int ch = 0; ch < nchunks; ++ch) {
    const int t0 = ch * 32;

    // S = Q * K^T for 32 keys (two 16-key WMMA column tiles)
    v8f s[2] = {};
#pragma unroll
    for (int j = 0; j < 2; ++j) {
      const __bf16* kr = Kp + (size_t)(t0 + 16 * j + ln) * Dc;
#pragma unroll
      for (int kk = 0; kk < 4; ++kk) {
        v8bf lo = *(const v8bf*)&kr[kk * 32 + 8 * hl];
        v8bf hi = *(const v8bf*)&kr[kk * 32 + 16 + 8 * hl];
        s[j] = wmma_bf16(qa[kk], cat16(lo, hi), s[j]);
      }
    }

    // scale + causal mask + online softmax update; emit P (bf16) via LDS
#pragma unroll
    for (int v = 0; v < 8; ++v) {
      const int qi = q0 + v + 8 * hl;
      float x0 = s[0][v] * scale2;
      float x1 = s[1][v] * scale2;
      if (t0 + ln > qi)      x0 = -INFINITY;
      if (t0 + 16 + ln > qi) x1 = -INFINITY;
      float mx = fmaxf(x0, x1);
#pragma unroll
      for (int off = 8; off; off >>= 1) mx = fmaxf(mx, __shfl_xor(mx, off, 32));
      float mnew = fmaxf(m_i[v], mx);
      float p0 = exp2f(x0 - mnew);
      float p1 = exp2f(x1 - mnew);
      float alpha = exp2f(m_i[v] - mnew);
      float rs = p0 + p1;
#pragma unroll
      for (int off = 8; off; off >>= 1) rs += __shfl_xor(rs, off, 32);
      l_i[v] = l_i[v] * alpha + rs;
      m_i[v] = mnew;
#pragma unroll
      for (int dt = 0; dt < 8; ++dt) o[dt][v] *= alpha;
      pl[(v + 8 * hl) * PLD + ln] = (__bf16)p0;
      pl[(v + 8 * hl) * PLD + 16 + ln] = (__bf16)p1;
    }

    // reload P in A-fragment layout (same-wave DS ops are in-order)
    v8bf plo = *(const v8bf*)&pl[ln * PLD + 8 * hl];
    v8bf phi = *(const v8bf*)&pl[ln * PLD + 16 + 8 * hl];
    v16bf pa = cat16(plo, phi);

    // O += P * V  (contract over the 32 keys; Vt gives K-contiguous B frags)
#pragma unroll
    for (int dt = 0; dt < 8; ++dt) {
      const __bf16* vr = Vp + (size_t)(dt * 16 + ln) * Sc + t0;
      v8bf lo = *(const v8bf*)&vr[8 * hl];
      v8bf hi = *(const v8bf*)&vr[16 + 8 * hl];
      o[dt] = wmma_bf16(pa, cat16(lo, hi), o[dt]);
    }
  }

  // normalize and write r[B,S,E] (bf16) for the output projection
#pragma unroll
  for (int v = 0; v < 8; ++v) {
    const float inv = 1.0f / l_i[v];
    const int qi = q0 + v + 8 * hl;
    __bf16* rr = rb + ((size_t)b * Sc + qi) * Ec + hd * Dc;
#pragma unroll
    for (int dt = 0; dt < 8; ++dt) rr[dt * 16 + ln] = (__bf16)(o[dt][v] * inv);
  }
}

// ---------------------------------------------------------------------------
// Launch: convert -> QKV GEMM (scatter) -> flash attention -> out GEMM
// ---------------------------------------------------------------------------
extern "C" void kernel_launch(void* const* d_in, const int* in_sizes, int n_in,
                              void* d_out, int out_size, void* d_ws, size_t ws_size,
                              hipStream_t stream) {
  (void)in_sizes; (void)n_in; (void)out_size; (void)ws_size;
  const float* x    = (const float*)d_in[0];  // [B,S,E]
  const float* Wqkv = (const float*)d_in[1];  // [E,3E]
  const float* Wout = (const float*)d_in[2];  // [E,E]
  float* out = (float*)d_out;                 // [B,S,E]

  char* ws = (char*)d_ws;
  size_t off = 0;
  auto carve = [&](size_t elems) { void* p = ws + off; off += elems * sizeof(__bf16); return p; };
  const size_t nX = (size_t)Bc * Sc * Ec;          // 16M
  __bf16* xb    = (__bf16*)carve(nX);
  __bf16* Wqkvt = (__bf16*)carve((size_t)3 * Ec * Ec);
  __bf16* Woutt = (__bf16*)carve((size_t)Ec * Ec);
  __bf16* Qb    = (__bf16*)carve(nX);
  __bf16* Kb    = (__bf16*)carve(nX);
  __bf16* Vtb   = (__bf16*)carve(nX);
  __bf16* rb    = (__bf16*)carve(nX);

  cvt_f32_bf16<<<(int)(nX / 1024), 256, 0, stream>>>(x, xb, (int)nX);
  transpose_cvt<<<dim3(3 * Ec / 32, Ec / 32), dim3(32, 8), 0, stream>>>(Wqkv, Wqkvt, Ec, 3 * Ec);
  transpose_cvt<<<dim3(Ec / 32, Ec / 32), dim3(32, 8), 0, stream>>>(Wout, Woutt, Ec, Ec);

  // qkv = xb @ Wqkv  : M=8192, N=6144, K=2048 -> scatter to Q/K/Vt
  gemm_bf16<0><<<dim3(3 * Ec / 128, Bc * Sc / 128), 256, 0, stream>>>(
      xb, Wqkvt, nullptr, Qb, Kb, Vtb, Bc * Sc, 3 * Ec, Ec);

  flash_attn<<<dim3(Sc / 128, Bc * Hc), 256, 0, stream>>>(Qb, Kb, Vtb, rb);

  // out = r @ Wout : M=8192, N=2048, K=2048
  gemm_bf16<1><<<dim3(Ec / 128, Bc * Sc / 128), 256, 0, stream>>>(
      rb, Woutt, out, nullptr, nullptr, nullptr, Bc * Sc, Ec, Ec);
}
